// SpikeLoss_81776177316348
// MI455X (gfx1250) — matrix operand order repaired and checked by reading, not compiled, not run on hardware
//
#include <hip/hip_runtime.h>
#include <hip/hip_bf16.h>

// SpikeLoss: delta = outputs - psp(target); loss = 0.5*sum(delta^2)
// psp recast as triangular matmul over time: PSP = X[pix,128] @ W[128,128],
// W[s,t] = (s<=t, s,t<100) ? 0.2*0.8^(t-s) : 0.  (tau_s = 5 -> decay = 0.8)
// Memory-bound: ~105 MB read @ 23.3 TB/s => ~4.5us floor. WMMA bf16 handles
// the scan math; LDS staging fixes coalescing of the T-contiguous layout.
// W matrix staged via gfx1250 async global->LDS DMA (ASYNCcnt), overlapped
// with the target tile convert/stage.

#define TAU_INV 0.2f
#define DECAY   0.8f

constexpr int T   = 100;   // time steps
constexpr int TP  = 128;   // padded T
constexpr int PITCH = 132; // LDS row pitch in ushorts (bank-conflict pad)
constexpr int PIX_PER_BLOCK = 64;  // 4 waves x 16 pixels
constexpr int BLOCK = 128;         // 4 wave32s
constexpr int W_CHUNKS = (TP * PITCH * 2) / 16;  // 2112 x 16B

typedef __attribute__((ext_vector_type(16))) __bf16 v16bf;
typedef __attribute__((ext_vector_type(8)))  float  v8f;

union Frag { v16bf v; unsigned u[8]; };

__device__ __forceinline__ unsigned short f2bf(float f) {
    unsigned u = __float_as_uint(f);
    unsigned r = u + 0x7FFFu + ((u >> 16) & 1u);   // round-to-nearest-even
    return (unsigned short)(r >> 16);
}

// Build bf16 filter matrix Wt[t][s] (pitch 132) in workspace; zero the output.
__global__ void build_W(unsigned short* __restrict__ gW, float* __restrict__ out) {
    if (threadIdx.x == 0 && blockIdx.x == 0) out[0] = 0.0f;
    for (int idx = threadIdx.x; idx < TP * PITCH; idx += blockDim.x) {
        int t = idx / PITCH;
        int s = idx % PITCH;
        float v = 0.0f;
        if (s < T && t < T && s <= t)
            v = __powf(DECAY, (float)(t - s)) * TAU_INV;
        gW[idx] = f2bf(v);
    }
}

__global__ void __launch_bounds__(BLOCK)
spike_loss_main(const float* __restrict__ outputs,
                const float* __restrict__ target,
                const unsigned short* __restrict__ gW,
                float* __restrict__ out) {
    __shared__ __align__(16) unsigned short sWt[TP * PITCH];           // 33.0 KB
    __shared__ __align__(16) unsigned short sX[PIX_PER_BLOCK * PITCH]; // 16.5 KB
    __shared__ float sred[BLOCK];

    const int tid  = threadIdx.x;
    const int pix0 = blockIdx.x * PIX_PER_BLOCK;

    // Prefetch the outputs tile consumed after the WMMAs (global_prefetch_b8).
    __builtin_prefetch(&outputs[(size_t)(pix0 + (tid & 63)) * T + (tid >> 6) * 64], 0, 1);

    // --- Async DMA the filter matrix into LDS (ASYNCcnt path, no VGPR trip).
    // ISA: LDS address = generic addr[31:0]; vdst = LDS addr, vaddr = 64b global.
    for (int i = tid; i < W_CHUNKS; i += BLOCK) {
        unsigned           lds = (unsigned)(size_t)(sWt + i * 8);
        unsigned long long ga  = (unsigned long long)(size_t)(gW + i * 8);
        asm volatile("global_load_async_to_lds_b128 %0, %1, off"
                     :: "v"(lds), "v"(ga) : "memory");
    }

    // --- Overlap: stage target tile as bf16 (needs VALU convert), pad k>=100.
    for (int idx = tid; idx < PIX_PER_BLOCK * TP; idx += BLOCK) {
        int p = idx >> 7;       // pixel within block
        int k = idx & 127;      // time index
        float v = (k < T) ? target[(size_t)(pix0 + p) * T + k] : 0.0f;
        sX[p * PITCH + k] = f2bf(v);
    }

    asm volatile("s_wait_asynccnt 0" ::: "memory");
    __syncthreads();

    const int wv   = tid >> 5;        // wave id, owns pixels [wv*16, wv*16+16)
    const int ln   = tid & 31;
    const int mrow = ln & 15;         // A: matrix row m; B/C: column n
    const int hi   = ln >> 4;         // lane-group select

    float local = 0.0f;

    // 7 column tiles cover t = 0..111 (t >= 100 masked); tile 7 is all-invalid.
    // Fully unrolled so kcmax constant-folds: 16 static WMMAs, loads pipelined.
#pragma unroll
    for (int n = 0; n < 7; ++n) {
        v8f acc = {};
        const int kcmax = (n * 16 + 15) >> 5;   // skip structurally-zero k chunks
#pragma unroll
        for (int kc = 0; kc <= kcmax; ++kc) {
            Frag a, b;
#pragma unroll
            for (int v = 0; v < 8; ++v) {
                // A 16x32 bf16 layout: k = kc*32 + 2*(v&3) + (v>=4?16:0) + (hi?8:0)
                int ka = kc * 32 + 2 * (v & 3) + ((v >= 4) ? 16 : 0) + (hi ? 8 : 0);
                a.u[v] = *(const unsigned*)&sX[(wv * 16 + mrow) * PITCH + ka];
                // B 32x16 bf16 layout: k = kc*32 + 2*(v&3) + (v>=4?16:0) + (hi?8:0), n = ln&15
                int kb = kc * 32 + 2 * (v & 3) + ((v >= 4) ? 16 : 0) + (hi ? 8 : 0);
                b.u[v] = *(const unsigned*)&sWt[(n * 16 + mrow) * PITCH + kb];
            }
            acc = __builtin_amdgcn_wmma_f32_16x16x32_bf16(
                false, a.v, false, b.v, (short)0, acc, false, false);
        }
        // C/D layout: lane ln, VGPR r -> M = r + hi*8, N = ln&15
        const int tcol = n * 16 + mrow;
        if (tcol < T) {
#pragma unroll
            for (int r = 0; r < 8; ++r) {
                int p = pix0 + wv * 16 + r + hi * 8;
                float d = outputs[(size_t)p * T + tcol] - acc[r];
                local += 0.5f * d * d;
            }
        }
    }

    sred[tid] = local;
    __syncthreads();
#pragma unroll
    for (int s = BLOCK / 2; s > 0; s >>= 1) {
        if (tid < s) sred[tid] += sred[tid + s];
        __syncthreads();
    }
    if (tid == 0) atomicAdd(out, sred[0]);
}

extern "C" void kernel_launch(void* const* d_in, const int* in_sizes, int n_in,
                              void* d_out, int out_size, void* d_ws, size_t ws_size,
                              hipStream_t stream) {
    const float* outputs = (const float*)d_in[0];
    const float* target  = (const float*)d_in[1];
    float* out = (float*)d_out;
    unsigned short* gW = (unsigned short*)d_ws;   // 128*132*2 = 33,792 B

    const int npix   = in_sizes[0] / T;           // 131072
    const int nblock = npix / PIX_PER_BLOCK;      // 2048

    build_W<<<1, 256, 0, stream>>>(gW, out);
    spike_loss_main<<<nblock, BLOCK, 0, stream>>>(outputs, target, gW, out);
}
